// RGCN_558345749136
// MI455X (gfx1250) — matrix operand order repaired and checked by reading, not compiled, not run on hardware
//
#include <hip/hip_runtime.h>

#define NN 100000
#define RR 4
#define EE 500000
#define DIN 128
#define DHID 128
#define DOUT 64

typedef __attribute__((ext_vector_type(16))) __bf16 v16bf;
typedef __attribute__((ext_vector_type(8)))  __bf16 v8bf;
typedef __attribute__((ext_vector_type(8)))  float  v8f;

// Fire-and-forget native HBM float atomic (no return -> STOREcnt; drained by
// the implicit wait-idle at S_ENDPGM).
__device__ __forceinline__ void atomic_add_f32(float* p, float v) {
  asm volatile("global_atomic_add_f32 %0, %1, off" : : "v"(p), "v"(v) : "memory");
}

__global__ __launch_bounds__(256) void degree_kernel(const int* __restrict__ src,
                                                     const int* __restrict__ dst,
                                                     unsigned* __restrict__ outdeg,
                                                     unsigned* __restrict__ indeg,
                                                     int e) {
  int t = blockIdx.x * blockDim.x + threadIdx.x;
  if (t >= e) return;
  atomicAdd(&outdeg[src[t]], 1u);
  atomicAdd(&indeg[dst[t]], 1u);
}

// In-place uint degree -> rsqrt(max(deg,1)) norm.
__global__ __launch_bounds__(256) void norm_kernel(unsigned* __restrict__ deg, int n) {
  int t = blockIdx.x * blockDim.x + threadIdx.x;
  if (t >= n) return;
  float d = (float)deg[t];
  ((float*)deg)[t] = rsqrtf(fmaxf(d, 1.0f));
}

// One wave per edge; lane c moves float4 of the 128-wide feature row.
__global__ __launch_bounds__(256) void scatter_kernel(const float* __restrict__ X,
                                                      const int* __restrict__ src,
                                                      const int* __restrict__ dst,
                                                      const float* __restrict__ nsrc,
                                                      float* __restrict__ agg, int e) {
  int t = blockIdx.x * blockDim.x + threadIdx.x;
  int ei = t >> 5;
  if (ei >= e) return;
  int c = (t & 31) * 4;
  int s = src[ei];
  int d = dst[ei];
  float ns = nsrc[s];
  float4 v = *(const float4*)(X + (size_t)s * 128 + c);
  float* p = agg + (size_t)d * 128 + c;
  atomic_add_f32(p + 0, v.x * ns);
  atomic_add_f32(p + 1, v.y * ns);
  atomic_add_f32(p + 2, v.z * ns);
  atomic_add_f32(p + 3, v.w * ns);
}

// C[n,NCOL] (+)= norm_dst[row] * (A[n,128] @ W[128,NCOL]) via bf16 WMMA.
// Block = 8 waves -> 32 rows x 64 cols tile; each wave owns a 16x16 D tile.
// Requires: grid.x * 32 == n exactly (NN = 3125 * 32) -> no row guards.
// W staged in LDS pre-swizzled into B-fragment order:
//   Wl[(k>>4)*1536 + n*24 + (k&15)]  (48B row stride: conflict-free b128s).
// INIT: first relation (store), else accumulate.
// FINAL: last relation -> add sum_r bias[r][col] (+ ReLU if RELU).
template <int NCOL, bool INIT, bool FINAL, bool RELU>
__global__ __launch_bounds__(256) void gemm_accum_kernel(const float* __restrict__ A,
                                                         const float* __restrict__ Wg,
                                                         const float* __restrict__ ndst,
                                                         float* __restrict__ C,
                                                         const float* __restrict__ bias) {
  __shared__ __bf16 Wl[8 * 1536];  // 24 KB
  const int tid  = threadIdx.x;
  const int wave = tid >> 5;
  const int lane = tid & 31;
  const int half = lane >> 4;
  const int l16  = lane & 15;
  const int mtile = blockIdx.x * 32 + (wave >> 2) * 16;
  const int nbase = blockIdx.y * 64;
  const int nloc  = (wave & 3) * 16;  // local col tile in [0,64)

  // Cooperative swizzled W load: 1024 segments of 8 K-values, each packed to
  // a v8bf and stored with one ds_store_b128.
  for (int q = tid; q < 1024; q += 256) {
    int g   = q >> 7;        // k-group of 16 (0..7)
    int rem = q & 127;
    int c   = rem >> 1;      // column 0..63
    int hi  = rem & 1;       // which 8 of the 16 k's
    const float* wsrc = Wg + (size_t)(g * 16 + hi * 8) * NCOL + nbase + c;
    v8bf pk;
#pragma unroll
    for (int j = 0; j < 8; ++j) pk[j] = (__bf16)wsrc[(size_t)j * NCOL];
    *(v8bf*)&Wl[g * 1536 + c * 24 + hi * 8] = pk;
  }
  __syncthreads();

  const float* arow = A + (size_t)(mtile + l16) * 128;
  const __bf16* bbase = &Wl[(nloc + l16) * 24 + half * 8];

  v8f acc = {};
#pragma unroll
  for (int k0 = 0; k0 < 128; k0 += 32) {
    // A frag: 16-bit A 16x32 layout -> lane {half,l16}: K runs
    // [k0+8*half .. +7] and [k0+16+8*half .. +7], two aligned float4 pairs.
    const float4* p0 = (const float4*)(arow + k0 + half * 8);
    const float4* p1 = (const float4*)(arow + k0 + 16 + half * 8);
    float4 f0 = p0[0], f1 = p0[1], f2 = p1[0], f3 = p1[1];
    v16bf a;
    a[0]  = (__bf16)f0.x; a[1]  = (__bf16)f0.y; a[2]  = (__bf16)f0.z; a[3]  = (__bf16)f0.w;
    a[4]  = (__bf16)f1.x; a[5]  = (__bf16)f1.y; a[6]  = (__bf16)f1.z; a[7]  = (__bf16)f1.w;
    a[8]  = (__bf16)f2.x; a[9]  = (__bf16)f2.y; a[10] = (__bf16)f2.z; a[11] = (__bf16)f2.w;
    a[12] = (__bf16)f3.x; a[13] = (__bf16)f3.y; a[14] = (__bf16)f3.z; a[15] = (__bf16)f3.w;

    // B frag: K runs [k0+8*half .. +7] and [k0+16+8*half .. +7], matching A.
    const __bf16* wp = bbase + (k0 >> 4) * 1536;
    v8bf blo = *(const v8bf*)(wp);
    v8bf bhi = *(const v8bf*)(wp + 1536);
    v16bf b = __builtin_shufflevector(blo, bhi, 0, 1, 2, 3, 4, 5, 6, 7,
                                      8, 9, 10, 11, 12, 13, 14, 15);

    acc = __builtin_amdgcn_wmma_f32_16x16x32_bf16(false, a, false, b,
                                                  (short)0, acc, false, false);
  }

  // D layout: VGPR v <-> row mtile + v + 8*half, col nbase + nloc + l16.
  const int ncol = nbase + nloc + l16;
  const int rbase = mtile + 8 * half;

  float nd[8];
#pragma unroll
  for (int v = 0; v < 8; ++v) nd[v] = ndst[rbase + v];

  float bsum = 0.0f;
  if (FINAL) {
#pragma unroll
    for (int r = 0; r < RR; ++r) bsum += bias[r * NCOL + ncol];
  }

#pragma unroll
  for (int v = 0; v < 8; ++v) {
    float val = acc[v] * nd[v];
    float* cp = C + (size_t)(rbase + v) * NCOL + ncol;
    if (INIT) {
      *cp = val;
    } else {
      val += *cp;
      if (FINAL) {
        val += bsum;
        if (RELU) val = fmaxf(val, 0.0f);
      }
      *cp = val;
    }
  }
}

extern "C" void kernel_launch(void* const* d_in, const int* in_sizes, int n_in,
                              void* d_out, int out_size, void* d_ws, size_t ws_size,
                              hipStream_t stream) {
  (void)in_sizes; (void)n_in; (void)out_size; (void)ws_size;
  const float* x  = (const float*)d_in[0];
  const int* esrc = (const int*)d_in[1];
  const int* edst = (const int*)d_in[2];
  const float* W1 = (const float*)d_in[3];
  const float* b1 = (const float*)d_in[4];
  const float* W2 = (const float*)d_in[5];
  const float* b2 = (const float*)d_in[6];
  float* out = (float*)d_out;

  char* ws = (char*)d_ws;
  size_t off = 0;
  auto carve = [&](size_t bytes) {
    char* p = ws + off;
    off += (bytes + 255) & ~(size_t)255;
    return p;
  };
  unsigned* outdeg = (unsigned*)carve((size_t)RR * NN * 4);  // becomes norm_src
  unsigned* indeg  = (unsigned*)carve((size_t)RR * NN * 4);  // becomes norm_dst
  float* agg = (float*)carve((size_t)NN * DIN * 4);
  float* h   = (float*)carve((size_t)NN * DHID * 4);
  float* nsrc = (float*)outdeg;
  float* ndst = (float*)indeg;

  // Degrees + norms (recomputed every call: ws is not re-zeroed by harness).
  hipMemsetAsync(outdeg, 0, (size_t)RR * NN * 4, stream);
  hipMemsetAsync(indeg,  0, (size_t)RR * NN * 4, stream);
  for (int r = 0; r < RR; ++r) {
    degree_kernel<<<(EE + 255) / 256, 256, 0, stream>>>(
        esrc + (size_t)r * EE, edst + (size_t)r * EE,
        outdeg + (size_t)r * NN, indeg + (size_t)r * NN, EE);
  }
  norm_kernel<<<(RR * NN + 255) / 256, 256, 0, stream>>>(outdeg, RR * NN);
  norm_kernel<<<(RR * NN + 255) / 256, 256, 0, stream>>>(indeg,  RR * NN);

  const int scatter_grid = (EE * 32 + 255) / 256;
  const dim3 g1(NN / 32, DHID / 64);
  const dim3 g2(NN / 32, 1);

  // Layer 1: h = relu(sum_r nd_r * (agg_r @ W1_r) + sum_r b1_r).
  for (int r = 0; r < RR; ++r) {
    hipMemsetAsync(agg, 0, (size_t)NN * DIN * 4, stream);
    scatter_kernel<<<scatter_grid, 256, 0, stream>>>(
        x, esrc + (size_t)r * EE, edst + (size_t)r * EE,
        nsrc + (size_t)r * NN, agg, EE);
    const float* w = W1 + (size_t)r * DIN * DHID;
    const float* nd = ndst + (size_t)r * NN;
    if (r == 0)
      gemm_accum_kernel<DHID, true, false, false><<<g1, 256, 0, stream>>>(agg, w, nd, h, nullptr);
    else if (r == RR - 1)
      gemm_accum_kernel<DHID, false, true, true><<<g1, 256, 0, stream>>>(agg, w, nd, h, b1);
    else
      gemm_accum_kernel<DHID, false, false, false><<<g1, 256, 0, stream>>>(agg, w, nd, h, nullptr);
  }

  // Layer 2: out = sum_r nd_r * (agg_r @ W2_r) + sum_r b2_r.
  for (int r = 0; r < RR; ++r) {
    hipMemsetAsync(agg, 0, (size_t)NN * DHID * 4, stream);
    scatter_kernel<<<scatter_grid, 256, 0, stream>>>(
        h, esrc + (size_t)r * EE, edst + (size_t)r * EE,
        nsrc + (size_t)r * NN, agg, EE);
    const float* w = W2 + (size_t)r * DHID * DOUT;
    const float* nd = ndst + (size_t)r * NN;
    if (r == 0)
      gemm_accum_kernel<DOUT, true, false, false><<<g2, 256, 0, stream>>>(agg, w, nd, out, nullptr);
    else if (r == RR - 1)
      gemm_accum_kernel<DOUT, false, true, false><<<g2, 256, 0, stream>>>(agg, w, nd, out, b2);
    else
      gemm_accum_kernel<DOUT, false, false, false><<<g2, 256, 0, stream>>>(agg, w, nd, out, nullptr);
  }
}
